// SpikingQNetwork_43061342110009
// MI455X (gfx1250) — compile-verified, matrix operand order checked by reading, not run on hardware
//
#include <hip/hip_runtime.h>

typedef __attribute__((ext_vector_type(2)))  float    v2f;
typedef __attribute__((ext_vector_type(8)))  float    v8f;
typedef __attribute__((ext_vector_type(8)))  _Float16 v8h;
typedef __attribute__((ext_vector_type(16))) _Float16 v16h;
typedef __attribute__((ext_vector_type(4)))  unsigned int u32x4;
typedef __attribute__((ext_vector_type(8)))  int      i32x8;
typedef __attribute__((ext_vector_type(4)))  int      i32x4;

#define B_SZ  16384
#define D_SZ  512
#define H_SZ  2048
#define H2_SZ 1024
#define A_SZ  8
#define NSTEP 8

// ---------------------------------------------------------------------------
// Tensor Data Mover: 2D tile load Global -> LDS (D# per CDNA5 ISA 8.3/8.4).
// clang-23 / therock 6-arg builtin form.
// ---------------------------------------------------------------------------
__device__ __forceinline__ void tdm_load_2d(unsigned int lds_off, const void* gptr,
                                            unsigned int elem_log2,  // 0:1B 1:2B 2:4B
                                            unsigned int tile_d0, unsigned int tile_d1,
                                            unsigned int tensor_d0, unsigned int tensor_d1,
                                            unsigned int stride_d0) {
  unsigned long long ga = (unsigned long long)gptr;
  u32x4 g0;
  g0[0] = 1u;                                              // count=1, user desc
  g0[1] = lds_off;                                         // LDS byte address
  g0[2] = (unsigned int)ga;                                // global_addr[31:0]
  g0[3] = (unsigned int)((ga >> 32) & 0x1FFFFFFu) | (2u << 30);  // addr[56:32], type=2
  i32x8 g1;
  g1[0] = (int)(elem_log2 << 16);                          // wg_mask=0, data_size
  g1[1] = (int)((tensor_d0 & 0xFFFFu) << 16);              // tensor_dim0 lo16
  g1[2] = (int)(((tensor_d0 >> 16) & 0xFFFFu) | ((tensor_d1 & 0xFFFFu) << 16));
  g1[3] = (int)(((tensor_d1 >> 16) & 0xFFFFu) | ((tile_d0 & 0xFFFFu) << 16));
  g1[4] = (int)(tile_d1 & 0xFFFFu);                        // tile_dim1, tile_dim2=0
  g1[5] = (int)stride_d0;                                  // dim0 stride lo32
  g1[6] = 0;
  g1[7] = 0;
  i32x4 z4 = {};
  i32x8 z8 = {};
  __builtin_amdgcn_tensor_load_to_lds(g0, g1, z4, z4, z8, 0);
}

// two spike bytes (bit0 of byte0/byte1 of x) -> two packed fp16 {0,1}
__device__ __forceinline__ unsigned int pair16(unsigned int x) {
  return ((x & 1u) | ((x & 0x100u) << 8)) * 0x3C00u;
}
// two adjacent bits (bit0,bit1 of b) -> two packed fp16 {0,1}
__device__ __forceinline__ unsigned int pb(unsigned int b) {
  return ((b & 1u) | ((b & 2u) << 15)) * 0x3C00u;
}
// lane-varying select of one of 8 wave-uniform values (7 cndmask tree)
__device__ __forceinline__ unsigned int sel8(const unsigned int b[8], int i) {
  unsigned int x0 = (i & 1) ? b[1] : b[0];
  unsigned int x1 = (i & 1) ? b[3] : b[2];
  unsigned int x2 = (i & 1) ? b[5] : b[4];
  unsigned int x3 = (i & 1) ? b[7] : b[6];
  unsigned int y0 = (i & 2) ? x1 : x0;
  unsigned int y1 = (i & 2) ? x3 : x2;
  return (i & 4) ? y1 : y0;
}

// ---------------------------------------------------------------------------
// K0: W2 (1024x2048 fp32) -> fp16; Wo (8x1024) -> fp16 padded to 16 rows.
// ---------------------------------------------------------------------------
__global__ void k0_prep(const float* __restrict__ W2, _Float16* __restrict__ W2h,
                        const float* __restrict__ Wo, _Float16* __restrict__ Woh) {
  int i = blockIdx.x * blockDim.x + threadIdx.x;
  if (i < H2_SZ * H_SZ) W2h[i] = (_Float16)W2[i];
  if (i < 16 * H2_SZ) {
    int a = i >> 10, n = i & (H2_SZ - 1);
    Woh[i] = (a < A_SZ) ? (_Float16)Wo[a * H2_SZ + n] : (_Float16)0.0f;
  }
}

// ---------------------------------------------------------------------------
// K1: cur1 = x @ W1^T + b1 via fp32 WMMA; 8-step lif1 recursion in registers;
// emit bit-packed spikes.  8 waves/block share one TDM-staged 16-row x tile.
// ---------------------------------------------------------------------------
__global__ void k1_gemm1_spikes(const float* __restrict__ x,
                                const float* __restrict__ W1,
                                const float* __restrict__ b1,
                                const float* __restrict__ beta1,
                                unsigned char* __restrict__ spk) {
  __shared__ float lds_x[16 * D_SZ];                       // 32 KB
  const int lane = threadIdx.x & 31;
  const int wave = threadIdx.x >> 5;
  const int l15  = lane & 15;
  const int hi   = lane >> 4;

  const int tilesN = H_SZ / 16;                            // 128 (multiple of 8)
  int tile0 = blockIdx.x * 8;
  int m0 = (tile0 / tilesN) * 16;
  int n0 = ((tile0 % tilesN) + wave) * 16;

  if (wave == 0) {
    tdm_load_2d((unsigned int)(size_t)lds_x, x + (size_t)m0 * D_SZ,
                2 /*4B*/, D_SZ, 16, D_SZ, B_SZ, D_SZ);
    __builtin_amdgcn_s_wait_tensorcnt(0);
  }
  __syncthreads();

  const float* ap = lds_x + l15 * D_SZ + 2 * hi;
  const float* bp = W1 + (size_t)(n0 + l15) * D_SZ + 2 * hi;

  v8f c = {};
#pragma unroll 8
  for (int kk = 0; kk < D_SZ; kk += 4) {
    v2f a = *(const v2f*)(ap + kk);
    v2f b = *(const v2f*)(bp + kk);
    c = __builtin_amdgcn_wmma_f32_16x16x4_f32(false, a, false, b,
                                              (short)0, c, false, false);
  }

  float bias = b1[n0 + l15];
  float be1  = fminf(fmaxf(*beta1, 0.0f), 1.0f);

#pragma unroll
  for (int r = 0; r < 8; ++r) {
    float cur = c[r] + bias;
    float m = 0.0f;
    unsigned int bits = 0;
#pragma unroll
    for (int t = 0; t < NSTEP; ++t) {
      float reset = (m > 1.0f) ? 1.0f : 0.0f;
      m = be1 * m + cur - reset;                 // THR = 1
      if (m > 1.0f) bits |= (1u << t);
    }
    int row = m0 + r + 8 * hi;
    spk[(size_t)row * H_SZ + n0 + l15] = (unsigned char)bits;
  }
}

// ---------------------------------------------------------------------------
// K2: fused 8-step lif2 + li_out partials.  Per wave: 16x32 tile, 2 passes x
// 4 per-step accumulators (A fragments from TDM-staged LDS feed 2 WMMAs).
// Epilogue: ballot-packed spk2 -> one f16 WMMA per step against padded Wo.
// ---------------------------------------------------------------------------
__global__ void k2_steps(const unsigned char* __restrict__ spk,
                         const _Float16* __restrict__ W2h,
                         const _Float16* __restrict__ Woh,
                         const float* __restrict__ b2,
                         const float* __restrict__ beta2,
                         float* __restrict__ pstep4) {
  __shared__ unsigned char lds_spk[16 * H_SZ];             // 32 KB
  __shared__ float pout[NSTEP][16][A_SZ];                  //  4 KB
  const int lane = threadIdx.x & 31;
  const int wave = threadIdx.x >> 5;
  const int l15  = lane & 15;
  const int hi   = lane >> 4;

  int tm = blockIdx.x >> 2;                                // 1024 row tiles
  int nb = blockIdx.x & 3;                                 // 4 N-slices
  int m0 = tm * 16;
  int n0 = nb * 256 + wave * 32;

  for (int i = threadIdx.x; i < NSTEP * 16 * A_SZ; i += 256)
    ((float*)pout)[i] = 0.0f;
  if (wave == 0) {
    tdm_load_2d((unsigned int)(size_t)lds_spk, spk + (size_t)m0 * H_SZ,
                0 /*1B*/, H_SZ, 16, H_SZ, B_SZ, H_SZ);
    __builtin_amdgcn_s_wait_tensorcnt(0);
  }
  __syncthreads();

  const unsigned long long* lsp =
      (const unsigned long long*)(lds_spk + l15 * H_SZ);
  const _Float16* wp0 = W2h + (size_t)(n0 + l15) * H_SZ;
  const _Float16* wp1 = W2h + (size_t)(n0 + 16 + l15) * H_SZ;

  float b2n0 = b2[n0 + l15];
  float b2n1 = b2[n0 + 16 + l15];
  float be2  = fminf(fmaxf(*beta2, 0.0f), 1.0f);

  // B fragment for the Wo output GEMM (constant per wave): lane = action col
  v16h bfo;
  {
    const _Float16* wrow = Woh + (size_t)l15 * H2_SZ + n0;
    v8h o0 = *(const v8h*)(wrow + 8 * hi);
    v8h o1 = *(const v8h*)(wrow + 16 + 8 * hi);
#pragma unroll
    for (int i = 0; i < 8; ++i) { bfo[i] = o0[i]; bfo[8 + i] = o1[i]; }
  }

  float mem2[2][8];
#pragma unroll
  for (int r = 0; r < 8; ++r) { mem2[0][r] = 0.0f; mem2[1][r] = 0.0f; }

  for (int tg = 0; tg < 2; ++tg) {                         // steps 4*tg .. 4*tg+3
    v8f acc[4][2];
#pragma unroll
    for (int tt = 0; tt < 4; ++tt) { acc[tt][0] = (v8f){}; acc[tt][1] = (v8f){}; }

    for (int kk = 0; kk < H_SZ; kk += 32) {
      unsigned long long s0 = lsp[(kk >> 3) + hi];
      unsigned long long s1 = lsp[(kk >> 3) + hi + 2];

      v16h bf0, bf1;
      {
        v8h a0 = *(const v8h*)(wp0 + kk + 8 * hi);
        v8h a1 = *(const v8h*)(wp0 + kk + 16 + 8 * hi);
        v8h c0 = *(const v8h*)(wp1 + kk + 8 * hi);
        v8h c1 = *(const v8h*)(wp1 + kk + 16 + 8 * hi);
#pragma unroll
        for (int i = 0; i < 8; ++i) {
          bf0[i] = a0[i]; bf0[8 + i] = a1[i];
          bf1[i] = c0[i]; bf1[8 + i] = c1[i];
        }
      }
      if (kk + 512 < H_SZ) {
        __builtin_prefetch(wp0 + kk + 512, 0, 0);
        __builtin_prefetch(wp1 + kk + 512, 0, 0);
      }

#pragma unroll
      for (int tt = 0; tt < 4; ++tt) {
        const int t = tg * 4 + tt;
        union { v16h h; unsigned int u[8]; } af;
        unsigned long long s0t = s0 >> t, s1t = s1 >> t;
        unsigned int x0 = (unsigned int)s0t, x1 = (unsigned int)(s0t >> 32);
        unsigned int y0 = (unsigned int)s1t, y1 = (unsigned int)(s1t >> 32);
        af.u[0] = pair16(x0);        af.u[1] = pair16(x0 >> 16);
        af.u[2] = pair16(x1);        af.u[3] = pair16(x1 >> 16);
        af.u[4] = pair16(y0);        af.u[5] = pair16(y0 >> 16);
        af.u[6] = pair16(y1);        af.u[7] = pair16(y1 >> 16);
        acc[tt][0] = __builtin_amdgcn_wmma_f32_16x16x32_f16(
            false, af.h, false, bf0, (short)0, acc[tt][0], false, false);
        acc[tt][1] = __builtin_amdgcn_wmma_f32_16x16x32_f16(
            false, af.h, false, bf1, (short)0, acc[tt][1], false, false);
      }
    }

    // lif2 recursion + ballot-packed spk2, then one WMMA vs Wo per step
#pragma unroll
    for (int tt = 0; tt < 4; ++tt) {
      const int t = tg * 4 + tt;
      unsigned int bal0[8], bal1[8];
#pragma unroll
      for (int r = 0; r < 8; ++r) {
        float rst0 = (mem2[0][r] > 1.0f) ? 1.0f : 0.0f;
        mem2[0][r] = be2 * mem2[0][r] + acc[tt][0][r] + b2n0 - rst0;
        bal0[r] = (unsigned int)__ballot(mem2[0][r] > 1.0f);
        float rst1 = (mem2[1][r] > 1.0f) ? 1.0f : 0.0f;
        mem2[1][r] = be2 * mem2[1][r] + acc[tt][1][r] + b2n1 - rst1;
        bal1[r] = (unsigned int)__ballot(mem2[1][r] > 1.0f);
      }
      // rebuild this lane's A fragment (row = l15) from the 16 uniform masks
      unsigned int sh = 8 * hi + ((l15 & 8) ? 16u : 0u);
      unsigned int mA = sel8(bal0, l15 & 7) >> sh;   // n_local 0..15  (elems 0-7)
      unsigned int mB = sel8(bal1, l15 & 7) >> sh;   // n_local 16..31 (elems 8-15)
      union { v16h h; unsigned int u[8]; } af;
      af.u[0] = pb(mA);      af.u[1] = pb(mA >> 2);
      af.u[2] = pb(mA >> 4); af.u[3] = pb(mA >> 6);
      af.u[4] = pb(mB);      af.u[5] = pb(mB >> 2);
      af.u[6] = pb(mB >> 4); af.u[7] = pb(mB >> 6);
      v8f o = (v8f){};
      o = __builtin_amdgcn_wmma_f32_16x16x32_f16(false, af.h, false, bfo,
                                                 (short)0, o, false, false);
      // C layout: lane = action col (valid a < 8), VGPR r = row r + 8*hi
#pragma unroll
      for (int r = 0; r < 8; ++r)
        if (l15 < A_SZ) atomicAdd(&pout[t][r + 8 * hi][l15], o[r]);  // ds_add_f32
    }
  }
  __syncthreads();

  // plain (non-atomic) store of this block's partial into its N-slice
  for (int i = threadIdx.x; i < NSTEP * 16 * A_SZ; i += 256) {
    int t = i / (16 * A_SZ);
    int rem = i - t * 16 * A_SZ;
    int r = rem >> 3, a = rem & 7;
    pstep4[(((size_t)t * B_SZ + m0 + r) * A_SZ + a) * 4 + nb] = pout[t][r][a];
  }
}

// ---------------------------------------------------------------------------
// K3: sum the 4 N-slice partials, apply li_out recursion + running max.
// ---------------------------------------------------------------------------
__global__ void k3_out(const float* __restrict__ pstep4,
                       const float* __restrict__ bo,
                       const float* __restrict__ beta_out,
                       float* __restrict__ out) {
  int i = blockIdx.x * blockDim.x + threadIdx.x;           // row*8 + a
  if (i >= B_SZ * A_SZ) return;
  int a = i & 7;
  int row = i >> 3;
  float beo = fminf(fmaxf(*beta_out, 0.0f), 1.0f);
  float boa = bo[a];
  float mo = 0.0f, mx = -1e9f;
#pragma unroll
  for (int t = 0; t < NSTEP; ++t) {
    const float4 q = *(const float4*)(pstep4 +
        (((size_t)t * B_SZ + row) * A_SZ + a) * 4);
    mo = beo * mo + (q.x + q.y) + (q.z + q.w) + boa;
    mx = fmaxf(mx, mo);
  }
  out[i] = mx;
}

// ---------------------------------------------------------------------------
extern "C" void kernel_launch(void* const* d_in, const int* in_sizes, int n_in,
                              void* d_out, int out_size, void* d_ws, size_t ws_size,
                              hipStream_t stream) {
  const float* x        = (const float*)d_in[0];
  const float* W1       = (const float*)d_in[1];
  const float* b1       = (const float*)d_in[2];
  const float* W2       = (const float*)d_in[3];
  const float* b2       = (const float*)d_in[4];
  const float* Wo       = (const float*)d_in[5];
  const float* bo       = (const float*)d_in[6];
  const float* beta1    = (const float*)d_in[7];
  const float* beta2    = (const float*)d_in[8];
  const float* beta_out = (const float*)d_in[9];

  char* ws = (char*)d_ws;
  _Float16*      W2h    = (_Float16*)ws;                         //  4 MB
  unsigned char* spk    = (unsigned char*)(ws + (4ull << 20));   // 32 MB
  float*         pstep4 = (float*)(ws + (36ull << 20));          // 16 MB
  _Float16*      Woh    = (_Float16*)(ws + (52ull << 20));       // 32 KB

  k0_prep<<<(H2_SZ * H_SZ + 255) / 256, 256, 0, stream>>>(W2, W2h, Wo, Woh);

  k1_gemm1_spikes<<<(B_SZ / 16) * (H_SZ / 16) / 8, 256, 0, stream>>>(
      x, W1, b1, beta1, spk);

  k2_steps<<<(B_SZ / 16) * (H2_SZ / 256), 256, 0, stream>>>(
      spk, W2h, Woh, b2, beta2, pstep4);

  k3_out<<<(B_SZ * A_SZ + 255) / 256, 256, 0, stream>>>(
      pstep4, bo, beta_out, (float*)d_out);
}
